// Decoder_26199300506117
// MI455X (gfx1250) — compile-verified, matrix operand order checked by reading, not run on hardware
//
#include <hip/hip_runtime.h>
#include <hip/hip_bf16.h>

typedef __attribute__((ext_vector_type(16))) _Float16 v16h;
typedef __attribute__((ext_vector_type(8)))  _Float16 v8h;
typedef __attribute__((ext_vector_type(8)))  float    v8f;

#define B_      256
#define TCAP    51
#define TDEC    50
#define VOC     10000
#define VOC_PAD 10048   // next multiple of 64
#define WD_     512
#define H_      512
#define IC_     2048
#define G3      1536    // 3*H

__device__ __forceinline__ float sigm(float x) { return 1.0f / (1.0f + __expf(-x)); }

__device__ __forceinline__ v16h ldfrag(const _Float16* p) {
  v8h lo = *(const v8h*)p;
  v8h hi = *(const v8h*)(p + 16);
  return __builtin_shufflevector(lo, hi, 0,1,2,3,4,5,6,7,8,9,10,11,12,13,14,15);
}

// ---------------------------------------------------------------------------
// Sort-by-length (descending, stable): rank via O(B^2) count, B=256, 1 block.
// ---------------------------------------------------------------------------
__global__ void k_rank(const int* __restrict__ cap_lens, int* __restrict__ order,
                       int* __restrict__ lens) {
  int i  = threadIdx.x;
  int li = cap_lens[i];
  int r  = 0;
  for (int p = 0; p < B_; ++p) {
    int lp = cap_lens[p];
    r += (lp > li) || (lp == li && p < i);
  }
  order[r] = i;
  lens[r]  = li - 1;
}

// ---------------------------------------------------------------------------
// Gather sorted captions + image codes (f16) and emit caps/lens/order outputs.
// ---------------------------------------------------------------------------
__global__ void k_gather(const int* __restrict__ captions, const float* __restrict__ image_code,
                         const int* __restrict__ order, const int* __restrict__ lens,
                         int* __restrict__ caps_s, _Float16* __restrict__ img_h,
                         float* __restrict__ out_caps, float* __restrict__ out_lens,
                         float* __restrict__ out_order) {
  int tid = blockIdx.x * blockDim.x + threadIdx.x;
  if (tid >= B_ * IC_) return;
  int b = tid / IC_, j = tid % IC_;
  int src = order[b];
  img_h[(size_t)b * IC_ + j] = (_Float16)image_code[(size_t)src * IC_ + j];
  if (j < TCAP) {
    int c = captions[src * TCAP + j];
    caps_s[b * TCAP + j]   = c;
    out_caps[b * TCAP + j] = (float)c;
  }
  if (j == 0) {
    out_lens[b]  = (float)lens[b];
    out_order[b] = (float)order[b];
  }
}

// ---------------------------------------------------------------------------
// f32 -> f16 conversion (grid-stride).
// ---------------------------------------------------------------------------
__global__ void k_cvt(const float* __restrict__ s, _Float16* __restrict__ d, size_t n) {
  size_t i      = (size_t)blockIdx.x * blockDim.x + threadIdx.x;
  size_t stride = (size_t)gridDim.x * blockDim.x;
  for (; i < n; i += stride) d[i] = (_Float16)s[i];
}

// f32 -> f16 with zero padding past nrows_src rows (row length = H_).
__global__ void k_cvt_pad(const float* __restrict__ s, _Float16* __restrict__ d,
                          int nrows_src, int nrows_dst) {
  size_t n = (size_t)nrows_dst * H_;
  size_t i = (size_t)blockIdx.x * blockDim.x + threadIdx.x;
  size_t stride = (size_t)gridDim.x * blockDim.x;
  for (; i < n; i += stride) {
    int row = (int)(i / H_);
    d[i] = (row < nrows_src) ? (_Float16)s[i] : (_Float16)0.0f;
  }
}

// ---------------------------------------------------------------------------
// Embedding gather for all timesteps: xs[(t*B+b), :] = f16(embed_w[caps[b,t]])
// ---------------------------------------------------------------------------
__global__ void k_embed(const int* __restrict__ caps_s, const float* __restrict__ embed_w,
                        _Float16* __restrict__ xs) {
  int tid = blockIdx.x * blockDim.x + threadIdx.x;
  if (tid >= TDEC * B_ * WD_) return;
  int j  = tid % WD_;
  int tb = tid / WD_;
  int b  = tb % B_;
  int t  = tb / B_;
  int idx = caps_s[b * TCAP + t];
  xs[(size_t)tb * WD_ + j] = (_Float16)embed_w[(size_t)idx * WD_ + j];
}

// ---------------------------------------------------------------------------
// Simple WMMA GEMM (1 wave per 16x16 tile) — used for the small latency-bound
// recurrence GEMMs where wave-count parallelism beats reuse.
// ---------------------------------------------------------------------------
__global__ void k_gemm(const _Float16* __restrict__ A, int lda,
                       const _Float16* __restrict__ W, int ldw,
                       const float* __restrict__ bias,
                       float* __restrict__ C, int ldc,
                       int M, int N, int K) {
  int gw   = (int)(((size_t)blockIdx.x * blockDim.x + threadIdx.x) >> 5);
  int lane = threadIdx.x & 31;
  int mt_cnt = M >> 4, nt_cnt = N >> 4;
  if (gw >= mt_cnt * nt_cnt) return;
  int mt = gw / nt_cnt, nt = gw % nt_cnt;
  int row = lane & 15;
  int kb  = (lane >> 4) << 3;
  const _Float16* a = A + (size_t)(mt * 16 + row) * lda + kb;
  const _Float16* w = W + (size_t)(nt * 16 + row) * ldw + kb;
  v8f acc = {};
  #pragma unroll 4
  for (int k = 0; k < K; k += 32) {
    v16h av = ldfrag(a + k);
    v16h bv = ldfrag(w + k);
    acc = __builtin_amdgcn_wmma_f32_16x16x32_f16(false, av, false, bv, (short)0, acc,
                                                 false, false);
  }
  int n  = nt * 16 + (lane & 15);
  int mb = mt * 16 + ((lane >> 4) << 3);
  float bb = bias ? bias[n] : 0.0f;
  #pragma unroll
  for (int e = 0; e < 8; ++e) C[(size_t)(mb + e) * ldc + n] = acc[e] + bb;
}

// ---------------------------------------------------------------------------
// Register-blocked WMMA GEMM: 32x64 output per wave (2x4 fragments, 8 WMMAs
// per K-step, 12 b128 loads) -> ~21 FLOP/byte of L2 traffic. Npad must be a
// multiple of 64 with all W rows < Npad loadable; stores guarded at Nvalid.
// ---------------------------------------------------------------------------
__global__ void k_gemm64(const _Float16* __restrict__ A, int lda,
                         const _Float16* __restrict__ W, int ldw,
                         const float* __restrict__ bias,
                         float* __restrict__ C, int ldc,
                         int M, int Npad, int Nvalid, int K) {
  int gw   = (int)(((size_t)blockIdx.x * blockDim.x + threadIdx.x) >> 5);
  int lane = threadIdx.x & 31;
  int mt_cnt = M >> 5, nt_cnt = Npad >> 6;
  if (gw >= mt_cnt * nt_cnt) return;
  int mt = gw / nt_cnt, nt = gw % nt_cnt;
  int row = lane & 15;
  int kb  = (lane >> 4) << 3;
  const _Float16* a0 = A + (size_t)(mt * 32 + row) * lda + kb;
  const _Float16* a1 = a0 + (size_t)16 * lda;
  const _Float16* w0 = W + (size_t)(nt * 64 + row) * ldw + kb;
  size_t wstep = (size_t)16 * ldw;
  v8f acc[2][4] = {};
  #pragma unroll 2
  for (int k = 0; k < K; k += 32) {
    v16h av0 = ldfrag(a0 + k);
    v16h av1 = ldfrag(a1 + k);
    v16h bv0 = ldfrag(w0 + k);
    v16h bv1 = ldfrag(w0 + wstep + k);
    v16h bv2 = ldfrag(w0 + 2 * wstep + k);
    v16h bv3 = ldfrag(w0 + 3 * wstep + k);
    acc[0][0] = __builtin_amdgcn_wmma_f32_16x16x32_f16(false, av0, false, bv0, (short)0, acc[0][0], false, false);
    acc[0][1] = __builtin_amdgcn_wmma_f32_16x16x32_f16(false, av0, false, bv1, (short)0, acc[0][1], false, false);
    acc[0][2] = __builtin_amdgcn_wmma_f32_16x16x32_f16(false, av0, false, bv2, (short)0, acc[0][2], false, false);
    acc[0][3] = __builtin_amdgcn_wmma_f32_16x16x32_f16(false, av0, false, bv3, (short)0, acc[0][3], false, false);
    acc[1][0] = __builtin_amdgcn_wmma_f32_16x16x32_f16(false, av1, false, bv0, (short)0, acc[1][0], false, false);
    acc[1][1] = __builtin_amdgcn_wmma_f32_16x16x32_f16(false, av1, false, bv1, (short)0, acc[1][1], false, false);
    acc[1][2] = __builtin_amdgcn_wmma_f32_16x16x32_f16(false, av1, false, bv2, (short)0, acc[1][2], false, false);
    acc[1][3] = __builtin_amdgcn_wmma_f32_16x16x32_f16(false, av1, false, bv3, (short)0, acc[1][3], false, false);
  }
  int nlo = lane & 15;
  int mb  = (lane >> 4) << 3;
  #pragma unroll
  for (int i = 0; i < 2; ++i) {
    #pragma unroll
    for (int j = 0; j < 4; ++j) {
      int n = nt * 64 + j * 16 + nlo;
      if (n < Nvalid) {
        float bb = bias ? bias[n] : 0.0f;
        int mrow = mt * 32 + i * 16 + mb;
        #pragma unroll
        for (int e = 0; e < 8; ++e)
          C[(size_t)(mrow + e) * ldc + n] = acc[i][j][e] + bb;
      }
    }
  }
}

// ---------------------------------------------------------------------------
// gi0_all[(t*B+b), :] += gi_img[b, :]
// ---------------------------------------------------------------------------
__global__ void k_addb(float* __restrict__ gi_all, const float* __restrict__ gi_img) {
  size_t tid = (size_t)blockIdx.x * blockDim.x + threadIdx.x;
  size_t n   = (size_t)TDEC * B_ * G3;
  if (tid >= n) return;
  size_t per = (size_t)B_ * G3;
  gi_all[tid] += gi_img[tid % per];
}

// ---------------------------------------------------------------------------
// Split h0init (B x 2H) into layer-0 / layer-1 hidden (f32 + f16 copies).
// ---------------------------------------------------------------------------
__global__ void k_split(const float* __restrict__ h0init, float* __restrict__ h0,
                        float* __restrict__ h1, _Float16* __restrict__ h0h,
                        _Float16* __restrict__ h1h) {
  int tid = blockIdx.x * blockDim.x + threadIdx.x;
  if (tid >= B_ * 2 * H_) return;
  int b = tid / (2 * H_);
  int c = tid % (2 * H_);
  float v = h0init[tid];
  int l = c / H_, j = c % H_;
  if (l == 0) { h0[b * H_ + j] = v; h0h[b * H_ + j] = (_Float16)v; }
  else        { h1[b * H_ + j] = v; h1h[b * H_ + j] = (_Float16)v; }
}

// ---------------------------------------------------------------------------
// GRU gating: r,z,n + masked hidden update.
// ---------------------------------------------------------------------------
__global__ void k_gate(const float* __restrict__ gi, const float* __restrict__ gh,
                       const float* hprev, const int* __restrict__ lens, int t,
                       float* hout, _Float16* __restrict__ hout_h,
                       _Float16* __restrict__ hraw_h) {
  int tid = blockIdx.x * blockDim.x + threadIdx.x;
  if (tid >= B_ * H_) return;
  int b = tid / H_;
  int j = tid % H_;
  size_t g = (size_t)b * G3 + j;
  float r  = sigm(gi[g] + gh[g]);
  float z  = sigm(gi[g + H_] + gh[g + H_]);
  float nn = tanhf(gi[g + 2 * H_] + r * gh[g + 2 * H_]);
  float hp = hprev[tid];
  float hl = (1.0f - z) * nn + z * hp;
  hraw_h[tid] = (_Float16)hl;
  float hm = (t < lens[b]) ? hl : hp;
  hout[tid]   = hm;
  hout_h[tid] = (_Float16)hm;
}

// ---------------------------------------------------------------------------
// Vocab projection, register-blocked 32x64 per wave, with length mask and
// (t*B+b) -> (b,t) transpose in the epilogue. W is zero-padded to VOC_PAD.
// ---------------------------------------------------------------------------
__global__ void k_fc64(const _Float16* __restrict__ A /* (T*B) x H */,
                       const _Float16* __restrict__ W /* VOC_PAD x H */,
                       const float* __restrict__ bias /* VOC */,
                       const int* __restrict__ lens,
                       float* __restrict__ C /* B x T x V */) {
  int gw   = (int)(((size_t)blockIdx.x * blockDim.x + threadIdx.x) >> 5);
  int lane = threadIdx.x & 31;
  const int mt_cnt = (TDEC * B_) >> 5, nt_cnt = VOC_PAD >> 6;
  if (gw >= mt_cnt * nt_cnt) return;
  int mt = gw / nt_cnt, nt = gw % nt_cnt;
  int row = lane & 15;
  int kb  = (lane >> 4) << 3;
  const _Float16* a0 = A + (size_t)(mt * 32 + row) * H_ + kb;
  const _Float16* a1 = a0 + (size_t)16 * H_;
  const _Float16* w0 = W + (size_t)(nt * 64 + row) * H_ + kb;
  const size_t wstep = (size_t)16 * H_;
  v8f acc[2][4] = {};
  #pragma unroll 2
  for (int k = 0; k < H_; k += 32) {
    v16h av0 = ldfrag(a0 + k);
    v16h av1 = ldfrag(a1 + k);
    v16h bv0 = ldfrag(w0 + k);
    v16h bv1 = ldfrag(w0 + wstep + k);
    v16h bv2 = ldfrag(w0 + 2 * wstep + k);
    v16h bv3 = ldfrag(w0 + 3 * wstep + k);
    acc[0][0] = __builtin_amdgcn_wmma_f32_16x16x32_f16(false, av0, false, bv0, (short)0, acc[0][0], false, false);
    acc[0][1] = __builtin_amdgcn_wmma_f32_16x16x32_f16(false, av0, false, bv1, (short)0, acc[0][1], false, false);
    acc[0][2] = __builtin_amdgcn_wmma_f32_16x16x32_f16(false, av0, false, bv2, (short)0, acc[0][2], false, false);
    acc[0][3] = __builtin_amdgcn_wmma_f32_16x16x32_f16(false, av0, false, bv3, (short)0, acc[0][3], false, false);
    acc[1][0] = __builtin_amdgcn_wmma_f32_16x16x32_f16(false, av1, false, bv0, (short)0, acc[1][0], false, false);
    acc[1][1] = __builtin_amdgcn_wmma_f32_16x16x32_f16(false, av1, false, bv1, (short)0, acc[1][1], false, false);
    acc[1][2] = __builtin_amdgcn_wmma_f32_16x16x32_f16(false, av1, false, bv2, (short)0, acc[1][2], false, false);
    acc[1][3] = __builtin_amdgcn_wmma_f32_16x16x32_f16(false, av1, false, bv3, (short)0, acc[1][3], false, false);
  }
  int nlo = lane & 15;
  int mb  = (lane >> 4) << 3;
  #pragma unroll
  for (int i = 0; i < 2; ++i) {
    #pragma unroll
    for (int j = 0; j < 4; ++j) {
      int n = nt * 64 + j * 16 + nlo;
      if (n < VOC) {
        float bb = bias[n];
        #pragma unroll
        for (int e = 0; e < 8; ++e) {
          int m = mt * 32 + i * 16 + mb + e;
          int t = m >> 8;          // m / B_
          int b = m & (B_ - 1);
          float v = (t < lens[b]) ? (acc[i][j][e] + bb) : 0.0f;
          C[(size_t)b * TDEC * VOC + (size_t)t * VOC + n] = v;
        }
      }
    }
  }
}

// ---------------------------------------------------------------------------
extern "C" void kernel_launch(void* const* d_in, const int* in_sizes, int n_in,
                              void* d_out, int out_size, void* d_ws, size_t ws_size,
                              hipStream_t stream) {
  (void)in_sizes; (void)n_in; (void)out_size; (void)ws_size;

  const float* image_code = (const float*)d_in[0];
  const int*   captions   = (const int*)d_in[1];
  const int*   cap_lens   = (const int*)d_in[2];
  const float* embed_w    = (const float*)d_in[3];
  const float* init_w     = (const float*)d_in[4];
  const float* init_b     = (const float*)d_in[5];
  const float* w_ih0      = (const float*)d_in[6];
  const float* w_hh0      = (const float*)d_in[7];
  const float* b_ih0      = (const float*)d_in[8];
  const float* b_hh0      = (const float*)d_in[9];
  const float* w_ih1      = (const float*)d_in[10];
  const float* w_hh1      = (const float*)d_in[11];
  const float* b_ih1      = (const float*)d_in[12];
  const float* b_hh1      = (const float*)d_in[13];
  const float* fc_w       = (const float*)d_in[14];
  const float* fc_b       = (const float*)d_in[15];

  float* out_preds = (float*)d_out;                       // (B,T,V)
  float* out_caps  = out_preds + (size_t)B_ * TDEC * VOC; // (B,51)
  float* out_lens  = out_caps + B_ * TCAP;                // (B,)
  float* out_order = out_lens + B_;                       // (B,)

  // ---- workspace bump allocator (256-B aligned chunks) ----
  char* p = (char*)d_ws;
  auto alloc = [&](size_t bytes) -> void* {
    void* r = p; p += (bytes + 255) & ~(size_t)255; return r;
  };
  int*      ws_order = (int*)alloc(B_ * 4);
  int*      ws_lens  = (int*)alloc(B_ * 4);
  int*      ws_caps  = (int*)alloc((size_t)B_ * TCAP * 4);
  _Float16* img_h    = (_Float16*)alloc((size_t)B_ * IC_ * 2);
  _Float16* wih0h    = (_Float16*)alloc((size_t)G3 * (IC_ + WD_) * 2);
  _Float16* whh0h    = (_Float16*)alloc((size_t)G3 * H_ * 2);
  _Float16* wih1h    = (_Float16*)alloc((size_t)G3 * H_ * 2);
  _Float16* whh1h    = (_Float16*)alloc((size_t)G3 * H_ * 2);
  _Float16* initwh   = (_Float16*)alloc((size_t)2 * H_ * IC_ * 2);
  _Float16* fcwh     = (_Float16*)alloc((size_t)VOC_PAD * H_ * 2);
  _Float16* xs_h     = (_Float16*)alloc((size_t)TDEC * B_ * WD_ * 2);
  float*    h0init   = (float*)alloc((size_t)B_ * 2 * H_ * 4);
  float*    h0       = (float*)alloc((size_t)B_ * H_ * 4);
  float*    h1       = (float*)alloc((size_t)B_ * H_ * 4);
  _Float16* h0h      = (_Float16*)alloc((size_t)B_ * H_ * 2);
  _Float16* h1h      = (_Float16*)alloc((size_t)B_ * H_ * 2);
  _Float16* hl0h     = (_Float16*)alloc((size_t)B_ * H_ * 2);
  float*    gh0      = (float*)alloc((size_t)B_ * G3 * 4);
  float*    gi1      = (float*)alloc((size_t)B_ * G3 * 4);
  float*    gh1      = (float*)alloc((size_t)B_ * G3 * 4);
  float*    gi_img   = (float*)alloc((size_t)B_ * G3 * 4);
  _Float16* h1all    = (_Float16*)alloc((size_t)TDEC * B_ * H_ * 2);
  // 78.6 MB of precomputed layer-0 input gates live inside the (later
  // overwritten) prediction region of d_out — fully consumed before k_fc64.
  float* gi0_all = out_preds;

  // ---- sort + gather ----
  k_rank<<<1, B_, 0, stream>>>(cap_lens, ws_order, ws_lens);
  {
    int n = B_ * IC_;
    k_gather<<<(n + 255) / 256, 256, 0, stream>>>(captions, image_code, ws_order, ws_lens,
                                                  ws_caps, img_h, out_caps, out_lens,
                                                  out_order);
  }

  // ---- weight conversions to f16 ----
  auto cvt = [&](const float* s, _Float16* d, size_t n) {
    size_t blocks = (n + 255) / 256;
    if (blocks > 8192) blocks = 8192;
    k_cvt<<<(unsigned)blocks, 256, 0, stream>>>(s, d, n);
  };
  cvt(w_ih0,  wih0h,  (size_t)G3 * (IC_ + WD_));
  cvt(w_hh0,  whh0h,  (size_t)G3 * H_);
  cvt(w_ih1,  wih1h,  (size_t)G3 * H_);
  cvt(w_hh1,  whh1h,  (size_t)G3 * H_);
  cvt(init_w, initwh, (size_t)2 * H_ * IC_);
  k_cvt_pad<<<8192, 256, 0, stream>>>(fc_w, fcwh, VOC, VOC_PAD);

  // ---- embeddings for all timesteps ----
  {
    int n = TDEC * B_ * WD_;
    k_embed<<<(n + 255) / 256, 256, 0, stream>>>(ws_caps, embed_w, xs_h);
  }

  auto gemm = [&](const _Float16* A, int lda, const _Float16* W, int ldw,
                  const float* bias, float* C, int ldc, int M, int N, int K) {
    long waves   = (long)(M / 16) * (N / 16);
    long threads = waves * 32;
    k_gemm<<<(unsigned)((threads + 255) / 256), 256, 0, stream>>>(A, lda, W, ldw, bias, C,
                                                                  ldc, M, N, K);
  };
  auto gemm64 = [&](const _Float16* A, int lda, const _Float16* W, int ldw,
                    const float* bias, float* C, int ldc, int M, int Npad, int Nvalid,
                    int K) {
    long waves   = (long)(M / 32) * (Npad / 64);
    long threads = waves * 32;
    k_gemm64<<<(unsigned)((threads + 255) / 256), 256, 0, stream>>>(A, lda, W, ldw, bias,
                                                                    C, ldc, M, Npad,
                                                                    Nvalid, K);
  };

  // ---- initial hidden: h0init = img_h @ init_w^T + init_b ----
  gemm64(img_h, IC_, initwh, IC_, init_b, h0init, 2 * H_, B_, 2 * H_, 2 * H_, IC_);
  k_split<<<(B_ * 2 * H_ + 255) / 256, 256, 0, stream>>>(h0init, h0, h1, h0h, h1h);

  // ---- precompute layer-0 input gates for ALL steps ----
  // embedding part (time-varying): one big (12800 x 1536 x 512) GEMM, +b_ih0
  gemm64(xs_h, WD_, wih0h + IC_, IC_ + WD_, b_ih0, gi0_all, G3, TDEC * B_, G3, G3, WD_);
  // image part (time-invariant): computed once, broadcast-added over t
  gemm64(img_h, IC_, wih0h, IC_ + WD_, nullptr, gi_img, G3, B_, G3, G3, IC_);
  {
    size_t n = (size_t)TDEC * B_ * G3;
    k_addb<<<(unsigned)((n + 255) / 256), 256, 0, stream>>>(gi0_all, gi_img);
  }

  // ---- sequential recurrence: only the h-dependent GEMMs remain ----
  for (int t = 0; t < TDEC; ++t) {
    gemm(h0h, H_, whh0h, H_, b_hh0, gh0, G3, B_, G3, H_);
    k_gate<<<(B_ * H_ + 255) / 256, 256, 0, stream>>>(gi0_all + (size_t)t * B_ * G3, gh0,
                                                      h0, ws_lens, t, h0, h0h, hl0h);
    gemm(hl0h, H_, wih1h, H_, b_ih1, gi1, G3, B_, G3, H_);
    gemm(h1h,  H_, whh1h, H_, b_hh1, gh1, G3, B_, G3, H_);
    k_gate<<<(B_ * H_ + 255) / 256, 256, 0, stream>>>(gi1, gh1, h1, ws_lens, t, h1, h1h,
                                                      h1all + (size_t)t * B_ * H_);
  }

  // ---- one giant masked vocab projection: (12800 x 10048 x 512) ----
  {
    long waves   = (long)((TDEC * B_) / 32) * (VOC_PAD / 64);
    long threads = waves * 32;
    k_fc64<<<(unsigned)((threads + 255) / 256), 256, 0, stream>>>(h1all, fcwh, fc_b,
                                                                  ws_lens, out_preds);
  }
}
